// SparseMoE_Incremental_Learning_52561809768848
// MI455X (gfx1250) — compile-verified, matrix operand order checked by reading, not run on hardware
//
#include <hip/hip_runtime.h>

#define DEVI __device__ __forceinline__

// ---- problem sizes (match reference) ----
constexpr int BATCH = 4, SEQ = 1024, DIM = 512, NEXP = 10;
constexpr int HID   = 2048;            // 4*DIM
constexpr int CEDIM = 32, LINDIM = 928;
constexpr int NTOK  = BATCH * SEQ;     // 4096

// ---- GEMM tiling: block tile 64 x 128, K-step 32, 8 waves of 16x64 ----
constexpr int TM = 64, TN = 128, TK = 32;
constexpr int LSTR = TK + 8;           // padded LDS row stride (halfwords) -> 80B rows
constexpr int ABUF = TM * LSTR;        // elements per A buffer (double buffered)
constexpr int BBUF = TN * LSTR;        // elements per B buffer

typedef __attribute__((ext_vector_type(16))) __bf16 v16bf;
typedef __attribute__((ext_vector_type(8)))  float  v8f;
typedef __attribute__((ext_vector_type(4)))  unsigned int v4u;
typedef __attribute__((ext_vector_type(4)))  int v4i;
typedef __attribute__((ext_vector_type(8)))  int v8i;

struct alignas(16) U4 { unsigned int v[4]; };
struct U8x { U4 a, b; };
struct alignas(16) F4 { float v[4]; };

DEVI unsigned short f2bf(float f) {
  unsigned int u = __builtin_bit_cast(unsigned int, f);
  u += 0x7FFFu + ((u >> 16) & 1u);     // round-to-nearest-even
  return (unsigned short)(u >> 16);
}

DEVI float gelu_exact(float x) {
  return 0.5f * x * (1.0f + erff(x * 0.70710678118654752f));
}

// Build a 16-bit WMMA fragment from two 16B LDS chunks (K{koff..+7}, K{koff+16..+23})
DEVI v16bf ld_frag(const unsigned short* p0, const unsigned short* p1) {
  U8x q;
  q.a = *(const U4*)p0;
  q.b = *(const U4*)p1;
  return __builtin_bit_cast(v16bf, q);
}

// ---- CDNA5 async global->LDS copy (ASYNCcnt tracked, no VGPR round trip) ----
DEVI unsigned lds_addr32(const void* p) {     // flat LDS pointer: low 32 bits = LDS address
  return (unsigned)(uintptr_t)p;
}
DEVI void async_cp16(unsigned lds, const void* g) {
  asm volatile("global_load_async_to_lds_b128 %0, %1, off"
               :: "v"(lds), "v"((unsigned long long)(uintptr_t)g)
               : "memory");
}
DEVI void wait_async0() {
  asm volatile("s_wait_asynccnt 0x0" ::: "memory");
}

// ---- CDNA5 Tensor Data Mover: one instruction stages a whole 128x32 bf16 tile ----
#if __has_builtin(__builtin_amdgcn_tensor_load_to_lds) && __has_builtin(__builtin_amdgcn_s_wait_tensorcnt)
#define HAVE_TDM 1
#else
#define HAVE_TDM 0
#endif

#if HAVE_TDM
// 2-D tile: tile_dim0 = TK elems (64B rows), tile_dim1 = TN rows, row stride = stride_elems.
// LDS padding: 4 DWORDs after every 16 DWORDs -> 80B LDS rows (matches LSTR).
DEVI void tdm_load_2d(unsigned lds, const void* gptr, unsigned stride_elems) {
  unsigned long long ga = (unsigned long long)(uintptr_t)gptr;
  v4u g0;
  g0.x = 1u;                                   // count=1 valid descriptor, no gather
  g0.y = lds;                                  // LDS byte address (bits 63:32)
  g0.z = (unsigned)ga;                         // global_addr[31:0]  (bits 95:64)
  g0.w = (unsigned)(ga >> 32) | (2u << 30);    // global_addr[56:32] | type=2 ("image")
  v8i g1;
  g1[0] = (1 << 16)                            // data_size = 2 bytes
        | (1 << 20)                            // pad_enable
        | (3 << 22)                            // pad_interval: 16 DWORDs (64B)
        | (3 << 25);                           // pad_amount: 4 DWORDs (16B)
  g1[1] = (TK & 0xFFFF) << 16;                 // tensor_dim0 = 32 (atomic_barrier_addr = 0)
  g1[2] = ((unsigned)TK >> 16) | ((TN & 0xFFFF) << 16);  // tensor_dim1 = 128
  g1[3] = ((unsigned)TN >> 16) | (TK << 16);   // tile_dim0 = 32
  g1[4] = TN;                                  // tile_dim1 = 128, tile_dim2 = 0 (2D)
  g1[5] = (int)stride_elems;                   // tensor_dim0_stride[31:0]
  g1[6] = 0;                                   // stride0 hi | stride1 lo (unused, 2D)
  g1[7] = 0;
  v4i z4 = {0, 0, 0, 0};
#if __clang_major__ >= 23
  v8i z8 = {0, 0, 0, 0, 0, 0, 0, 0};
  __builtin_amdgcn_tensor_load_to_lds(g0, g1, z4, z4, z8, 0);
#else
  __builtin_amdgcn_tensor_load_to_lds(g0, g1, z4, z4, 0);
#endif
}
#endif

DEVI void wait_stage() {
#if HAVE_TDM
  __builtin_amdgcn_s_wait_tensorcnt((short)0);   // no-op for waves that issued no TDM
#endif
  wait_async0();
}

// one K-step of the 16x64-per-wave WMMA macro tile
DEVI void gemm_step(const unsigned short* Ac, const unsigned short* Bc,
                    int wm, int wn, int l15, int koff, v8f* acc) {
  const unsigned short* ar = Ac + (wm * 16 + l15) * LSTR + koff;
  v16bf af = ld_frag(ar, ar + 16);
  const unsigned short* b0 = Bc + (wn * 64 + l15) * LSTR + koff;
  v16bf bf0 = ld_frag(b0,             b0 + 16);
  v16bf bf1 = ld_frag(b0 + 16 * LSTR, b0 + 16 * LSTR + 16);
  v16bf bf2 = ld_frag(b0 + 32 * LSTR, b0 + 32 * LSTR + 16);
  v16bf bf3 = ld_frag(b0 + 48 * LSTR, b0 + 48 * LSTR + 16);
  acc[0] = __builtin_amdgcn_wmma_f32_16x16x32_bf16(false, af, false, bf0, (short)0, acc[0], false, false);
  acc[1] = __builtin_amdgcn_wmma_f32_16x16x32_bf16(false, af, false, bf1, (short)0, acc[1], false, false);
  acc[2] = __builtin_amdgcn_wmma_f32_16x16x32_bf16(false, af, false, bf2, (short)0, acc[2], false, false);
  acc[3] = __builtin_amdgcn_wmma_f32_16x16x32_bf16(false, af, false, bf3, (short)0, acc[3], false, false);
}

// ---- workspace layout (bytes, 256-aligned) ----
constexpr size_t O_CNT  = 0;                                      // NEXP ints
constexpr size_t O_TOK  = 256;                                    // NEXP*NTOK ints (slot=tok*2+k)
constexpr size_t O_GATE = O_TOK  + (size_t)NEXP * NTOK * 4;       // NEXP*NTOK floats
constexpr size_t O_XB   = O_GATE + (size_t)NEXP * NTOK * 4;       // x bf16 [N][D]
constexpr size_t O_W1T  = O_XB   + (size_t)NTOK * DIM * 2;        // W1^T bf16 [e][h][d]
constexpr size_t O_W2T  = O_W1T  + (size_t)NEXP * DIM * HID * 2;  // W2^T bf16 [e][d][h]
constexpr size_t O_HID  = O_W2T  + (size_t)NEXP * DIM * HID * 2;  // hid bf16 [N*2][H]
constexpr size_t O_YB   = O_HID  + (size_t)NTOK * 2 * HID * 2;    // y f32 [N*2][D]
// total ~92.3 MB

// ================= init (counts only; out fully rewritten by combine) =================
__global__ void moe_zero_cnt(int* cnt) {
  if (threadIdx.x < NEXP) cnt[threadIdx.x] = 0;
}

// ================= fp32 -> bf16 conversions =================
__global__ void moe_cvt_x(const float* __restrict__ x, unsigned short* __restrict__ xb, int n) {
  int i = blockIdx.x * blockDim.x + threadIdx.x;
  if (i < n) xb[i] = f2bf(x[i]);
}

// W1 (E,D,H) row-major  ->  w1t [e][h][d]
__global__ void moe_cvt_w1(const float* __restrict__ W1, unsigned short* __restrict__ w1t) {
  int i = blockIdx.x * blockDim.x + threadIdx.x;
  if (i >= NEXP * DIM * HID) return;
  int e = i / (DIM * HID);
  int r = i % (DIM * HID);
  int d = r / HID;
  int h = r % HID;
  w1t[((size_t)e * HID + h) * DIM + d] = f2bf(W1[i]);
}

// W2 (E,H,D) row-major  ->  w2t [e][d][h]
__global__ void moe_cvt_w2(const float* __restrict__ W2, unsigned short* __restrict__ w2t) {
  int i = blockIdx.x * blockDim.x + threadIdx.x;
  if (i >= NEXP * HID * DIM) return;
  int e = i / (HID * DIM);
  int r = i % (HID * DIM);
  int h = r / DIM;
  int d = r % DIM;
  w2t[((size_t)e * DIM + d) * HID + h] = f2bf(W2[i]);
}

// ================= router: one wave32 per token =================
__global__ __launch_bounds__(128)
void moe_router(const float* __restrict__ x, const float* __restrict__ dt,
                const float* __restrict__ dd, const float* __restrict__ drg,
                const float* __restrict__ den, const float* __restrict__ cemb,
                const float* __restrict__ Wg, const float* __restrict__ bg,
                const float* __restrict__ Wn, const float* __restrict__ bn,
                const float* __restrict__ noise, const int* __restrict__ city,
                float* __restrict__ gate1, int* __restrict__ cnt,
                int* __restrict__ tok_list, float* __restrict__ gate_list) {
  int wid  = threadIdx.x >> 5;
  int lane = threadIdx.x & 31;
  int n = blockIdx.x * (blockDim.x >> 5) + wid;
  if (n >= NTOK) return;
  int b = n / SEQ;

  const float* xr   = x   + (size_t)n * DIM;
  const float* dtr  = dt  + (size_t)n * (DIM / 4);
  const float* ddr  = dd  + (size_t)n * (DIM / 4);
  const float* drgr = drg + (size_t)n * (DIM / 8);
  const float* denr = den + (size_t)n * (DIM / 8);
  const float* cer  = cemb + (size_t)city[b] * CEDIM;

  float accG[NEXP], accN[NEXP];
#pragma unroll
  for (int e = 0; e < NEXP; ++e) { accG[e] = 0.f; accN[e] = 0.f; }

  for (int l = lane; l < LINDIM; l += 32) {   // 928 = 29*32, no remainder
    float hv;
    if      (l < 512) hv = xr[l];
    else if (l < 544) hv = cer[l - 512];
    else if (l < 672) hv = dtr[l - 544];
    else if (l < 800) hv = ddr[l - 672];
    else if (l < 864) hv = drgr[l - 800];
    else              hv = denr[l - 864];
#pragma unroll
    for (int e = 0; e < NEXP; ++e) {
      accG[e] = fmaf(hv, Wg[e * LINDIM + l], accG[e]);
      accN[e] = fmaf(hv, Wn[e * LINDIM + l], accN[e]);
    }
  }
#pragma unroll
  for (int e = 0; e < NEXP; ++e) {
#pragma unroll
    for (int m = 16; m >= 1; m >>= 1) {
      accG[e] += __shfl_xor(accG[e], m, 32);
      accN[e] += __shfl_xor(accN[e], m, 32);
    }
  }

  if (lane == 0) {
    float logits[NEXP], noisy[NEXP];
    float mx = -3.4e38f;
#pragma unroll
    for (int e = 0; e < NEXP; ++e) { logits[e] = accG[e] + bg[e]; mx = fmaxf(mx, logits[e]); }
    float s = 0.f, ex[NEXP];
#pragma unroll
    for (int e = 0; e < NEXP; ++e) { ex[e] = expf(logits[e] - mx); s += ex[e]; }
    float inv = 1.0f / s;
#pragma unroll
    for (int e = 0; e < NEXP; ++e) gate1[(size_t)n * NEXP + e] = ex[e] * inv;

#pragma unroll
    for (int e = 0; e < NEXP; ++e) {
      float nl = accN[e] + bn[e];
      float sp = (nl > 20.f) ? nl : log1pf(expf(nl));   // softplus
      noisy[e] = logits[e] + noise[(size_t)n * NEXP + e] * sp;
    }
    int i1 = 0;
#pragma unroll
    for (int e = 1; e < NEXP; ++e) if (noisy[e] > noisy[i1]) i1 = e;
    int i2 = (i1 == 0) ? 1 : 0;
#pragma unroll
    for (int e = 0; e < NEXP; ++e) if (e != i1 && noisy[e] > noisy[i2]) i2 = e;
    float m2 = fmaxf(noisy[i1], noisy[i2]);
    float e1 = expf(noisy[i1] - m2), e2 = expf(noisy[i2] - m2);
    float si = 1.0f / (e1 + e2);

    int p1 = atomicAdd(&cnt[i1], 1);
    tok_list[i1 * NTOK + p1]  = n * 2 + 0;
    gate_list[i1 * NTOK + p1] = e1 * si;
    int p2 = atomicAdd(&cnt[i2], 1);
    tok_list[i2 * NTOK + p2]  = n * 2 + 1;
    gate_list[i2 * NTOK + p2] = e2 * si;
  }
}

// ================= FFN1: hid = gelu(x @ W1 + b1) — async A, TDM B, double-buffered ====
__global__ __launch_bounds__(256)
void moe_ffn1(const unsigned short* __restrict__ xb, const unsigned short* __restrict__ w1t,
              const float* __restrict__ b1, unsigned short* __restrict__ hid,
              const int* __restrict__ cnt, const int* __restrict__ tok_list) {
  __shared__ __align__(16) unsigned short As[2 * ABUF];
  __shared__ __align__(16) unsigned short Bs[2 * BBUF];
  __shared__ int s_slot[TM];

  const int e  = blockIdx.x / (NTOK / TM);
  const int mt = blockIdx.x % (NTOK / TM);
  const int count = cnt[e];
  if (mt * TM >= count) return;
  const int n0 = blockIdx.y * TN;
  const int tid = threadIdx.x;

  if (tid < TM) {
    int a = mt * TM + tid;
    s_slot[tid] = (a < count) ? tok_list[e * NTOK + a] : -1;
  }
  __syncthreads();

  const int lane = tid & 31;
  const int wid  = tid >> 5;
  const int wm = wid & 3, wn = wid >> 2;      // 4x2 wave grid, each wave 16x64
  const int half = lane >> 4;
  const int l15  = lane & 15;
  const int koff = half ? 8 : 0;

  // A staging: 64 rows x 4 x 16B chunks, one async per thread; invalid rows clamp to tok 0
  const int rA = tid >> 2, ch = tid & 3;
  const int slotA = s_slot[rA];
  const int tokA  = (slotA >= 0) ? (slotA >> 1) : 0;
  const unsigned short* gA = xb + (size_t)tokA * DIM + ch * 8;
  const unsigned ldA = lds_addr32(As + rA * LSTR + ch * 8);

  // B staging: one TDM per tile (wave 0), or per-lane async fallback
  const unsigned short* gBtile = w1t + ((size_t)e * HID + n0) * DIM;
  const unsigned ldB = lds_addr32(Bs);
#if !HAVE_TDM
  const unsigned short* gB0 = gBtile + (size_t)rA * DIM + ch * 8;
  const unsigned short* gB1 = gB0 + (size_t)64 * DIM;
  const unsigned ldB0 = lds_addr32(Bs + rA * LSTR + ch * 8);
  const unsigned ldB1 = ldB0 + 64 * LSTR * 2;
#endif

  auto stage = [&](int sel, int kk) {
    async_cp16(ldA + sel * (ABUF * 2), gA + kk);
#if HAVE_TDM
    if (wid == 0) tdm_load_2d(ldB + sel * (BBUF * 2), gBtile + kk, DIM);
#else
    async_cp16(ldB0 + sel * (BBUF * 2), gB0 + kk);
    async_cp16(ldB1 + sel * (BBUF * 2), gB1 + kk);
#endif
  };

  v8f z8 = {0.f,0.f,0.f,0.f,0.f,0.f,0.f,0.f};
  v8f acc[4] = {z8, z8, z8, z8};

  stage(0, 0);
  const int nk = DIM / TK;
#pragma clang loop unroll(disable)
  for (int i = 0; i < nk - 1; ++i) {          // steady state: branch-free, no unroll ->
    wait_stage();                             // accumulators stay in-place across backedge
    __syncthreads();
    stage((i + 1) & 1, (i + 1) * TK);
    gemm_step(As + (i & 1) * ABUF, Bs + (i & 1) * BBUF, wm, wn, l15, koff, acc);
  }
  wait_stage();
  __syncthreads();
  gemm_step(As + ((nk - 1) & 1) * ABUF, Bs + ((nk - 1) & 1) * BBUF, wm, wn, l15, koff, acc);

  // epilogue: C/D layout -> VGPR v holds row v (lanes<16) / v+8 (lanes>=16), col = lane&15
#pragma unroll
  for (int v = 0; v < 8; ++v) {
    int r = wm * 16 + v + half * 8;
    int slot = s_slot[r];
    if (slot < 0) continue;
    unsigned short* hrow = hid + (size_t)slot * HID;
#pragma unroll
    for (int c = 0; c < 4; ++c) {
      int col = n0 + wn * 64 + c * 16 + l15;
      float f = acc[c][v] + b1[e * HID + col];
      hrow[col] = f2bf(gelu_exact(f));
    }
  }
}

// ================= FFN2: y[slot] = gate * (hid @ W2 + b2) ============================
__global__ __launch_bounds__(256)
void moe_ffn2(const unsigned short* __restrict__ hid, const unsigned short* __restrict__ w2t,
              const float* __restrict__ b2, float* __restrict__ yb,
              const int* __restrict__ cnt, const int* __restrict__ tok_list,
              const float* __restrict__ gate_list) {
  __shared__ __align__(16) unsigned short As[2 * ABUF];
  __shared__ __align__(16) unsigned short Bs[2 * BBUF];
  __shared__ int   s_slot[TM];
  __shared__ float s_gate[TM];

  const int e  = blockIdx.x / (NTOK / TM);
  const int mt = blockIdx.x % (NTOK / TM);
  const int count = cnt[e];
  if (mt * TM >= count) return;
  const int n0 = blockIdx.y * TN;
  const int tid = threadIdx.x;

  if (tid < TM) {
    int a = mt * TM + tid;
    bool ok = (a < count);
    s_slot[tid] = ok ? tok_list[e * NTOK + a] : -1;
    s_gate[tid] = ok ? gate_list[e * NTOK + a] : 0.f;
  }
  __syncthreads();

  const int lane = tid & 31;
  const int wid  = tid >> 5;
  const int wm = wid & 3, wn = wid >> 2;
  const int half = lane >> 4;
  const int l15  = lane & 15;
  const int koff = half ? 8 : 0;

  const int rA = tid >> 2, ch = tid & 3;
  const int slotA = s_slot[rA];
  const int srcSlot = (slotA >= 0) ? slotA : 0;
  const unsigned short* gA = hid + (size_t)srcSlot * HID + ch * 8;
  const unsigned ldA = lds_addr32(As + rA * LSTR + ch * 8);

  const unsigned short* gBtile = w2t + ((size_t)e * DIM + n0) * HID;
  const unsigned ldB = lds_addr32(Bs);
#if !HAVE_TDM
  const unsigned short* gB0 = gBtile + (size_t)rA * HID + ch * 8;
  const unsigned short* gB1 = gB0 + (size_t)64 * HID;
  const unsigned ldB0 = lds_addr32(Bs + rA * LSTR + ch * 8);
  const unsigned ldB1 = ldB0 + 64 * LSTR * 2;
#endif

  auto stage = [&](int sel, int kk) {
    async_cp16(ldA + sel * (ABUF * 2), gA + kk);
#if HAVE_TDM
    if (wid == 0) tdm_load_2d(ldB + sel * (BBUF * 2), gBtile + kk, HID);
#else
    async_cp16(ldB0 + sel * (BBUF * 2), gB0 + kk);
    async_cp16(ldB1 + sel * (BBUF * 2), gB1 + kk);
#endif
  };

  v8f z8 = {0.f,0.f,0.f,0.f,0.f,0.f,0.f,0.f};
  v8f acc[4] = {z8, z8, z8, z8};

  stage(0, 0);
  const int nk = HID / TK;
#pragma clang loop unroll(disable)
  for (int i = 0; i < nk - 1; ++i) {
    wait_stage();
    __syncthreads();
    stage((i + 1) & 1, (i + 1) * TK);
    gemm_step(As + (i & 1) * ABUF, Bs + (i & 1) * BBUF, wm, wn, l15, koff, acc);
  }
  wait_stage();
  __syncthreads();
  gemm_step(As + ((nk - 1) & 1) * ABUF, Bs + ((nk - 1) & 1) * BBUF, wm, wn, l15, koff, acc);

#pragma unroll
  for (int v = 0; v < 8; ++v) {
    int r = wm * 16 + v + half * 8;
    int slot = s_slot[r];
    if (slot < 0) continue;
    float g = s_gate[r];
    float* yrow = yb + (size_t)slot * DIM;
#pragma unroll
    for (int c = 0; c < 4; ++c) {
      int col = n0 + wn * 64 + c * 16 + l15;
      yrow[col] = (acc[c][v] + b2[e * DIM + col]) * g;
    }
  }
}

// ================= combine: out[n] = y[2n] + y[2n+1] (vectorized, no atomics) ========
__global__ void moe_combine(const float* __restrict__ yb, float* __restrict__ out) {
  int i = blockIdx.x * blockDim.x + threadIdx.x;     // over NTOK*DIM/4
  if (i >= NTOK * DIM / 4) return;
  int n = i >> 7;              // DIM/4 = 128
  int j = i & 127;
  const F4* a = (const F4*)(yb + (size_t)(2 * n)     * DIM) + j;
  const F4* b = (const F4*)(yb + (size_t)(2 * n + 1) * DIM) + j;
  F4 va = *a, vb = *b, r;
#pragma unroll
  for (int q = 0; q < 4; ++q) r.v[q] = va.v[q] + vb.v[q];
  ((F4*)out)[i] = r;
}

// ================= host launch =================
extern "C" void kernel_launch(void* const* d_in, const int* in_sizes, int n_in,
                              void* d_out, int out_size, void* d_ws, size_t ws_size,
                              hipStream_t stream) {
  (void)in_sizes; (void)n_in; (void)out_size; (void)ws_size;

  const float* x    = (const float*)d_in[0];
  const float* dt   = (const float*)d_in[1];
  const float* dd   = (const float*)d_in[2];
  const float* drg  = (const float*)d_in[3];
  const float* den  = (const float*)d_in[4];
  const float* cemb = (const float*)d_in[5];
  const float* Wg   = (const float*)d_in[6];
  const float* bg   = (const float*)d_in[7];
  const float* Wn   = (const float*)d_in[8];
  const float* bn   = (const float*)d_in[9];
  const float* W1   = (const float*)d_in[10];
  const float* b1   = (const float*)d_in[11];
  const float* W2   = (const float*)d_in[12];
  const float* b2   = (const float*)d_in[13];
  const float* noise = (const float*)d_in[14];
  const int*   city = (const int*)d_in[15];

  float* out   = (float*)d_out;                    // (B,T,D)
  float* gate1 = out + (size_t)NTOK * DIM;         // (B,T,E)

  char* ws = (char*)d_ws;
  int*            cnt       = (int*)(ws + O_CNT);
  int*            tok_list  = (int*)(ws + O_TOK);
  float*          gate_list = (float*)(ws + O_GATE);
  unsigned short* xb        = (unsigned short*)(ws + O_XB);
  unsigned short* w1t       = (unsigned short*)(ws + O_W1T);
  unsigned short* w2t       = (unsigned short*)(ws + O_W2T);
  unsigned short* hid       = (unsigned short*)(ws + O_HID);
  float*          yb        = (float*)(ws + O_YB);

  moe_zero_cnt<<<1, 32, 0, stream>>>(cnt);
  const int nX = NTOK * DIM;
  moe_cvt_x<<<(nX + 255) / 256, 256, 0, stream>>>(x, xb, nX);
  moe_cvt_w1<<<(NEXP * DIM * HID + 255) / 256, 256, 0, stream>>>(W1, w1t);
  moe_cvt_w2<<<(NEXP * HID * DIM + 255) / 256, 256, 0, stream>>>(W2, w2t);

  moe_router<<<NTOK / 4, 128, 0, stream>>>(x, dt, dd, drg, den, cemb,
                                           Wg, bg, Wn, bn, noise, city,
                                           gate1, cnt, tok_list, gate_list);

  moe_ffn1<<<dim3(NEXP * (NTOK / TM), HID / TN), 256, 0, stream>>>(
      xb, w1t, b1, hid, cnt, tok_list);

  moe_ffn2<<<dim3(NEXP * (NTOK / TM), DIM / TN), 256, 0, stream>>>(
      hid, w2t, b2, yb, cnt, tok_list, gate_list);

  moe_combine<<<(NTOK * DIM / 4 + 255) / 256, 256, 0, stream>>>(yb, out);
}
